// VectorQuantizer_4569845203156
// MI455X (gfx1250) — matrix operand order repaired and checked by reading, not compiled, not run on hardware
//
#include <hip/hip_runtime.h>
#include <cstddef>
#include <cstdint>

// ---------------------------------------------------------------------------
// VQ-VAE vector quantizer for MI455X (gfx1250, wave32, WMMA)
//   z:   (32, 256, 32, 32) f32   -> 32768 vectors of dim 256
//   emb: (1024, 256) f32 codebook
// Distance GEMM via split-bf16 (hi/lo) on V_WMMA_F32_16X16X32_BF16:
//   z.e = hi_z.hi_e + hi_z.lo_e + lo_z.hi_e   (f32 accumulate, ~2^-17 rel err)
// Block = 32-row M-tile, wave = 2 M-tiles x 4 N-tiles -> each B fragment in
// registers feeds 6 WMMAs (halves L2 codebook traffic vs 16-row blocks).
// Norms, argmin, z_q gather, loss all remain exact f32.
// ---------------------------------------------------------------------------

#define N_TOT    32768              // B*H*W flat vectors
#define N_E      1024               // codebook size
#define E_DIM    256                // code dimension
#define Z_ELEMS  (N_TOT * E_DIM)    // 8388608
#define R_BLK    32                 // rows per block (one full w-row)
#define A_STRIDE 36                 // padded LDS stride (u32) -> conflict-free

// d_out layout (floats, concatenated in reference return order):
#define OFF_LOSS 0
#define OFF_ZQ   1
#define OFF_PPL  (1 + Z_ELEMS)
#define OFF_CNT  (2 + Z_ELEMS)
#define OFF_IDX  (2 + Z_ELEMS + N_E)

typedef __attribute__((ext_vector_type(16))) __bf16 v16bf;
typedef __attribute__((ext_vector_type(8)))  float  v8f;

union Frag8 {
    uint32_t u[8];
    v16bf    v;
};

__device__ __forceinline__ uint32_t f2bf(float x) {
    // round-to-nearest-even f32 -> bf16 (inputs are finite)
    uint32_t u = __float_as_uint(x);
    return (u + 0x7FFFu + ((u >> 16) & 1u)) >> 16;
}
__device__ __forceinline__ float bf2f(uint32_t b) {
    return __uint_as_float(b << 16);
}

// ---------------------------------------------------------------------------
// Kernel 0: build K-pair-packed bf16 hi/lo transposed codebook
//   ETPhi/ETPlo[kpair][code] : u32 = bf16(K=2kp) | bf16(K=2kp+1)<<16
// plus exact ||e||^2, zero counts + loss accumulator.
// grid = 1024 blocks (one per code), block = 256 threads (one per dim).
// ---------------------------------------------------------------------------
__global__ __launch_bounds__(256) void vq_prep(const float* __restrict__ emb,
                                               uint32_t* __restrict__ ETPhi,
                                               uint32_t* __restrict__ ETPlo,
                                               float* __restrict__ enorm,
                                               float* __restrict__ lossacc,
                                               float* __restrict__ counts) {
    const int k = blockIdx.x;       // code index
    const int t = threadIdx.x;      // dim index
    const float v = emb[(size_t)k * E_DIM + t];

    const uint32_t hb = f2bf(v);
    const uint32_t lb = f2bf(v - bf2f(hb));
    const uint32_t hN = (uint32_t)__shfl_xor((int)hb, 1, 32);   // partner dim t^1
    const uint32_t lN = (uint32_t)__shfl_xor((int)lb, 1, 32);
    if ((t & 1) == 0) {
        ETPhi[(size_t)(t >> 1) * N_E + k] = hb | (hN << 16);
        ETPlo[(size_t)(t >> 1) * N_E + k] = lb | (lN << 16);
    }

    float s = v * v;
    #pragma unroll
    for (int off = 16; off >= 1; off >>= 1) s += __shfl_xor(s, off, 32);
    __shared__ float wsum[8];
    if ((t & 31) == 0) wsum[t >> 5] = s;
    __syncthreads();
    if (t == 0) {
        float tot = 0.f;
        #pragma unroll
        for (int i = 0; i < 8; ++i) tot += wsum[i];
        enorm[k] = tot;
        counts[k] = 0.f;
        if (k == 0) *lossacc = 0.f;
    }
}

// ---------------------------------------------------------------------------
// Kernel 1: main. One block per 32-row M-tile (one full (b,h) w-row, so all
// z / z_q global accesses are 128B-contiguous per wave). 512 threads =
// 16 waves; wave w owns codes [w*64, w*64+64) as 4 N-tiles, and both 16-row
// M-subtiles. K-loop: 8 steps of 32; per step 24 WMMAs, each B frag reused 6x.
// ---------------------------------------------------------------------------
__global__ __launch_bounds__(512) void vq_main(const float* __restrict__ z,
                                               const float* __restrict__ emb,
                                               const uint32_t* __restrict__ ETPhi,
                                               const uint32_t* __restrict__ ETPlo,
                                               const float* __restrict__ enorm,
                                               float* __restrict__ out,
                                               float* __restrict__ lossacc) {
    __shared__ float    A_lds[E_DIM * R_BLK];            // exact f32 z tile [c][i], 32KB
    __shared__ uint32_t Ahi[(E_DIM / 2) * A_STRIDE];     // packed bf16-hi [kp][i], 18KB
    __shared__ uint32_t Alo[(E_DIM / 2) * A_STRIDE];     // packed bf16-lo [kp][i], 18KB
    __shared__ float    en_lds[N_E];                     // ||e||^2, 4KB
    __shared__ float    sel_lds[R_BLK * 257];            // selected codebook rows (padded)
    __shared__ float    red_val[16 * R_BLK];             // per-wave argmin [wave][row]
    __shared__ int      red_idx[16 * R_BLK];
    __shared__ int      fin_idx[R_BLK];

    const int tid  = threadIdx.x;
    const int lane = tid & 31;
    const int wv   = tid >> 5;        // wave 0..15
    const int half = lane >> 4;       // 0: lanes 0-15, 1: lanes 16-31
    const int m    = lane & 15;

    // M-tile -> (b, h); z[b][c][h][w] flat = b*262144 + c*1024 + h*32 + w
    const int m0 = blockIdx.x * R_BLK;
    const int h  = blockIdx.x & 31;
    const int b  = blockIdx.x >> 5;
    const size_t zbase = (size_t)b * (E_DIM * 1024) + (size_t)h * 32;

    // --- cooperative load: f32 A tile (32 rows x 256 dims) + e_norms ------
    {
        const int i  = tid & 31;      // w position
        const int cg = tid >> 5;      // 0..15
        #pragma unroll
        for (int p = 0; p < 16; ++p) {
            const int c = cg + p * 16;
            A_lds[c * R_BLK + i] = z[zbase + (size_t)c * 1024 + i];
        }
        en_lds[tid]       = enorm[tid];
        en_lds[tid + 512] = enorm[tid + 512];
    }
    __syncthreads();

    // --- split z tile into packed bf16 hi/lo (K-pair-packed, padded) ------
    #pragma unroll
    for (int q = 0; q < 8; ++q) {
        const int idx = tid + q * 512;        // 0..4095
        const int kp  = idx >> 5;
        const int i   = idx & 31;
        const float v0 = A_lds[(2 * kp)     * R_BLK + i];
        const float v1 = A_lds[(2 * kp + 1) * R_BLK + i];
        const uint32_t h0 = f2bf(v0), h1 = f2bf(v1);
        Ahi[kp * A_STRIDE + i] = h0 | (h1 << 16);
        Alo[kp * A_STRIDE + i] = f2bf(v0 - bf2f(h0)) | (f2bf(v1 - bf2f(h1)) << 16);
    }
    __syncthreads();

    // --- distance GEMM: acc[mt][nt] = z_tile * E^T via split-bf16 WMMA ----
    v8f acc[2][4];
    #pragma unroll
    for (int mt = 0; mt < 2; ++mt)
        #pragma unroll
        for (int nt = 0; nt < 4; ++nt) acc[mt][nt] = (v8f){};

    const int n0 = wv * 64;
    #pragma unroll 2
    for (int k0 = 0; k0 < E_DIM; k0 += 32) {
        const int kb = k0 >> 1;               // K-pair base
        // A fragments: VGPR 0-3 -> K pairs at base(+half*4); 4-7 -> base+8(+half*4)
        Frag8 a_hi[2], a_lo[2];
        #pragma unroll
        for (int mt = 0; mt < 2; ++mt) {
            const int r = mt * 16 + m;
            #pragma unroll
            for (int j = 0; j < 4; ++j) {
                const int kpA = kb + half * 4 + j;
                a_hi[mt].u[j]     = Ahi[kpA * A_STRIDE + r];
                a_lo[mt].u[j]     = Alo[kpA * A_STRIDE + r];
                const int kpB = kb + 8 + half * 4 + j;
                a_hi[mt].u[j + 4] = Ahi[kpB * A_STRIDE + r];
                a_lo[mt].u[j + 4] = Alo[kpB * A_STRIDE + r];
            }
        }
        // B fragments: VGPR 0-3 -> K pairs at base(+half*4); 4-7 -> base+8(+half*4)
        const size_t brow = (size_t)(kb + half * 4) * N_E + n0 + m;
        const uint32_t* bph = ETPhi + brow;
        const uint32_t* bpl = ETPlo + brow;
        #pragma unroll
        for (int nt = 0; nt < 4; ++nt) {
            Frag8 b_hi, b_lo;
            #pragma unroll
            for (int j = 0; j < 8; ++j) {
                b_hi.u[j] = bph[(size_t)(j + (j & 4)) * N_E + nt * 16];
                b_lo.u[j] = bpl[(size_t)(j + (j & 4)) * N_E + nt * 16];
            }
            #pragma unroll
            for (int mt = 0; mt < 2; ++mt) {
                acc[mt][nt] = __builtin_amdgcn_wmma_f32_16x16x32_bf16(
                    false, a_hi[mt].v, false, b_hi.v, (short)0, acc[mt][nt], false, false);
                acc[mt][nt] = __builtin_amdgcn_wmma_f32_16x16x32_bf16(
                    false, a_hi[mt].v, false, b_lo.v, (short)0, acc[mt][nt], false, false);
                acc[mt][nt] = __builtin_amdgcn_wmma_f32_16x16x32_bf16(
                    false, a_lo[mt].v, false, b_hi.v, (short)0, acc[mt][nt], false, false);
            }
        }
    }

    // --- per-lane argmin over this wave's 64 codes, both M-subtiles -------
    // C layout: VGPR j, lanes 0-15 -> (M=j, N=lane); lanes 16-31 -> (M=j+8, N=lane-16)
    float bv[2][8];
    int   bi[2][8];
    #pragma unroll
    for (int mt = 0; mt < 2; ++mt)
        #pragma unroll
        for (int j = 0; j < 8; ++j) { bv[mt][j] = 3.4e38f; bi[mt][j] = 0; }
    #pragma unroll
    for (int nt = 0; nt < 4; ++nt) {
        const int n = n0 + nt * 16 + m;
        const float en = en_lds[n];
        #pragma unroll
        for (int mt = 0; mt < 2; ++mt) {
            const v8f av = acc[mt][nt];
            #pragma unroll
            for (int j = 0; j < 8; ++j) {
                const float v = en - 2.0f * av[j];
                if (v < bv[mt][j]) { bv[mt][j] = v; bi[mt][j] = n; }  // nt asc => first-min
            }
        }
    }
    // butterfly across the 16 lanes of each half (rows stay within halves)
    #pragma unroll
    for (int off = 8; off >= 1; off >>= 1) {
        #pragma unroll
        for (int mt = 0; mt < 2; ++mt) {
            #pragma unroll
            for (int j = 0; j < 8; ++j) {
                const float ov = __shfl_xor(bv[mt][j], off, 32);
                const int   oi = __shfl_xor(bi[mt][j], off, 32);
                if (ov < bv[mt][j] || (ov == bv[mt][j] && oi < bi[mt][j])) {
                    bv[mt][j] = ov; bi[mt][j] = oi;
                }
            }
        }
    }
    if (m == 0) {
        #pragma unroll
        for (int mt = 0; mt < 2; ++mt) {
            #pragma unroll
            for (int j = 0; j < 8; ++j) {
                const int row = mt * 16 + j + 8 * half;
                red_val[wv * R_BLK + row] = bv[mt][j];
                red_idx[wv * R_BLK + row] = bi[mt][j];
            }
        }
    }
    __syncthreads();

    // --- cross-wave argmin (16 waves x 32 rows), emit index + count -------
    if (tid < R_BLK) {
        float best = red_val[tid];
        int   bidx = red_idx[tid];
        for (int wq = 1; wq < 16; ++wq) {
            const float v  = red_val[wq * R_BLK + tid];
            const int   ix = red_idx[wq * R_BLK + tid];
            if (v < best || (v == best && ix < bidx)) { best = v; bidx = ix; }
        }
        fin_idx[tid] = bidx;
        out[OFF_IDX + m0 + tid] = (float)bidx;
        atomicAdd(&out[OFF_CNT + bidx], 1.0f);
    }
    __syncthreads();

    // --- stage the 32 selected codebook rows (exact f32) into LDS ----------
    #pragma unroll
    for (int g = 0; g < 2; ++g) {
        const int row = wv + g * 16;
        const int eidx = fin_idx[row];
        const float* erow = emb + (size_t)eidx * E_DIM;
        #pragma unroll
        for (int p = 0; p < 8; ++p) {
            const int c = lane + p * 32;
            sel_lds[row * 257 + c] = erow[c];
        }
    }
    __syncthreads();

    // --- scatter z_q back to (B,C,H,W) + loss partial ----------------------
    float lsum = 0.f;
    {
        const int i  = tid & 31;      // w position
        const int cg = tid >> 5;      // 0..15
        #pragma unroll
        for (int p = 0; p < 16; ++p) {
            const int c = cg + p * 16;
            const float e  = sel_lds[i * 257 + c];
            const float zv = A_lds[c * R_BLK + i];
            const float d  = e - zv;
            lsum += d * d;
            out[OFF_ZQ + zbase + (size_t)c * 1024 + i] = e;   // 128B contiguous / wave
        }
    }
    #pragma unroll
    for (int off = 16; off >= 1; off >>= 1) lsum += __shfl_xor(lsum, off, 32);
    if (lane == 0) atomicAdd(lossacc, lsum);
}

// ---------------------------------------------------------------------------
// Kernel 2: perplexity from counts, final loss scalar. One block, 1024 thr.
// ---------------------------------------------------------------------------
__global__ __launch_bounds__(1024) void vq_final(float* __restrict__ out,
                                                 const float* __restrict__ lossacc) {
    __shared__ float sbuf[32];
    const int t = threadIdx.x;
    const float c  = out[OFF_CNT + t];
    const float em = c * (1.0f / (float)N_TOT);
    float term = em * logf(em + 1e-10f);
    #pragma unroll
    for (int off = 16; off >= 1; off >>= 1) term += __shfl_xor(term, off, 32);
    if ((t & 31) == 0) sbuf[t >> 5] = term;
    __syncthreads();
    if (t == 0) {
        float s = 0.f;
        #pragma unroll
        for (int i = 0; i < 32; ++i) s += sbuf[i];
        out[OFF_PPL]  = expf(-s);
        // loss = mean((sg(z)-zq)^2) + 0.25*mean((z-sg(zq))^2) = 1.25*mean((z-zq)^2)
        out[OFF_LOSS] = 1.25f * (*lossacc) * (1.0f / (float)Z_ELEMS);
    }
}

// ---------------------------------------------------------------------------
extern "C" void kernel_launch(void* const* d_in, const int* in_sizes, int n_in,
                              void* d_out, int out_size, void* d_ws, size_t ws_size,
                              hipStream_t stream) {
    const float* z   = (const float*)d_in[0];   // (32,256,32,32)
    const float* emb = (const float*)d_in[1];   // (1024,256)
    float* out = (float*)d_out;

    uint32_t* ETPhi  = (uint32_t*)d_ws;                         // 128*1024 u32 (512KB)
    uint32_t* ETPlo  = ETPhi + (size_t)(E_DIM / 2) * N_E;       // 512KB
    float*    enorm  = (float*)(ETPlo + (size_t)(E_DIM / 2) * N_E);  // 4KB
    float*    lossacc = enorm + N_E;                            // 1 float

    vq_prep <<<N_E, E_DIM, 0, stream>>>(emb, ETPhi, ETPlo, enorm, lossacc, out + OFF_CNT);
    vq_main <<<N_TOT / R_BLK, 512, 0, stream>>>(z, emb, ETPhi, ETPlo, enorm, out, lossacc);
    vq_final<<<1, N_E, 0, stream>>>(out, lossacc);
}
